// GAT_86973087744576
// MI455X (gfx1250) — matrix operand order repaired and checked by reading, not compiled, not run on hardware
//
#include <hip/hip_runtime.h>
#include <math.h>

typedef __bf16 bf16;
typedef __attribute__((ext_vector_type(16))) __bf16 v16bf;
typedef __attribute__((ext_vector_type(8)))  __bf16 v8bf;
typedef __attribute__((ext_vector_type(8)))  float  v8f;

#define NEG_SLOPE 0.2f

// ---------------- helpers ----------------

__device__ __forceinline__ void edge_ends(const int* __restrict__ ei, int E, int e,
                                          int& s, int& d) {
  if (e < E) { s = ei[e]; d = ei[E + e]; }
  else       { s = e - E; d = s; }            // appended self loops
}

// float atomic max via monotonic int/uint encoding (init must be -inf)
__device__ __forceinline__ void atomicMaxF(float* addr, float val) {
  if (val >= 0.0f) atomicMax((int*)addr, __float_as_int(val));
  else             atomicMin((unsigned int*)addr, __float_as_uint(val));
}

__device__ __forceinline__ v16bf cat16(v8bf lo, v8bf hi) {
  v16bf v;
#pragma unroll
  for (int i = 0; i < 8; ++i) { v[i] = lo[i]; v[i + 8] = hi[i]; }
  return v;
}

// A fragment (16-bit A 16x32): lane half h: elems 0..7 = K(kc+8h..+7),
// elems 8..15 = K(kc+16+8h..+7)
__device__ __forceinline__ void load_afrag(const bf16* __restrict__ arow, int kc, int half,
                                           v8bf& lo, v8bf& hi) {
  lo = *(const v8bf*)(arow + kc + 8 * half);
  hi = *(const v8bf*)(arow + kc + 16 + 8 * half);
}

// B fragment (32x16): lane half h: elems 0..7 = K(kc+8h..+7),
// elems 8..15 = K(kc+16+8h..+7)  (B[k][n] = BT[n][k])
__device__ __forceinline__ void load_bfrag(const bf16* __restrict__ brow, int kc, int half,
                                           v8bf& lo, v8bf& hi) {
  lo = *(const v8bf*)(brow + kc + 8 * half);
  hi = *(const v8bf*)(brow + kc + 16 + 8 * half);
}

// ---------------- conversion kernels ----------------

__global__ void k_cvt_bf16(const float* __restrict__ in, bf16* __restrict__ out, size_t n) {
  size_t i = (size_t)blockIdx.x * blockDim.x + threadIdx.x;
  if (i < n) out[i] = (bf16)in[i];
}

// W [K, Nin] row-major  ->  WT [NoutPad, K] row-major, zero-padded columns
__global__ void k_transpose_cvt(const float* __restrict__ W, bf16* __restrict__ WT,
                                int K, int Nin, int NoutPad) {
  int i = blockIdx.x * blockDim.x + threadIdx.x;
  if (i >= NoutPad * K) return;
  int n = i / K, k = i - n * K;
  WT[i] = (n < Nin) ? (bf16)W[(size_t)k * Nin + n] : (bf16)0.0f;
}

__global__ void k_fill(float* __restrict__ p, float v, size_t n) {
  size_t i = (size_t)blockIdx.x * blockDim.x + threadIdx.x;
  if (i < n) p[i] = v;
}

// ---------------- WMMA GEMM: C[M,Npad] = A[M,K](bf16) * BT[Npad,K]^T ----------------
// One wave computes a 16 x (16*NT) strip with v_wmma_f32_16x16x32_bf16.
// Register double-buffered over K (requires K % 64 == 0, K >= 64): the WMMAs for
// chunk k issue while chunk k+32's global_load_b128s are still in flight.
template <int NT>
__global__ void k_gemm_wmma_bf16(const bf16* __restrict__ A,
                                 const bf16* __restrict__ BT,
                                 float* __restrict__ C,
                                 int M, int K, int Npad, int totalWaves) {
  int gtid = blockIdx.x * blockDim.x + threadIdx.x;
  int wave = gtid >> 5;
  if (wave >= totalWaves) return;
  int lane = gtid & 31;
  int l15  = lane & 15;
  int half = lane >> 4;
  int nGroups = Npad / (16 * NT);
  int mTile = wave / nGroups;
  int ng    = wave - mTile * nGroups;

  int mrow = mTile * 16 + l15;
  if (mrow >= M) mrow = M - 1;                 // ragged-M clamp (loads only)
  const bf16* arow = A + (size_t)mrow * K;
  const bf16* brow[NT];
#pragma unroll
  for (int j = 0; j < NT; ++j)
    brow[j] = BT + (size_t)((ng * NT + j) * 16 + l15) * K;

  v8f acc[NT] = {};
  v8bf a0lo, a0hi, a1lo, a1hi;
  v8bf b0lo[NT], b0hi[NT], b1lo[NT], b1hi[NT];

  // prologue: stage chunk 0
  load_afrag(arow, 0, half, a0lo, a0hi);
#pragma unroll
  for (int j = 0; j < NT; ++j) load_bfrag(brow[j], 0, half, b0lo[j], b0hi[j]);

  for (int kc = 0; kc < K - 64; kc += 64) {
    // stage kc+32, multiply kc
    load_afrag(arow, kc + 32, half, a1lo, a1hi);
#pragma unroll
    for (int j = 0; j < NT; ++j) load_bfrag(brow[j], kc + 32, half, b1lo[j], b1hi[j]);
    {
      v16bf av = cat16(a0lo, a0hi);
#pragma unroll
      for (int j = 0; j < NT; ++j)
        acc[j] = __builtin_amdgcn_wmma_f32_16x16x32_bf16(
            false, av, false, cat16(b0lo[j], b0hi[j]), (short)0, acc[j], false, false);
    }
    // stage kc+64, multiply kc+32
    load_afrag(arow, kc + 64, half, a0lo, a0hi);
#pragma unroll
    for (int j = 0; j < NT; ++j) load_bfrag(brow[j], kc + 64, half, b0lo[j], b0hi[j]);
    {
      v16bf av = cat16(a1lo, a1hi);
#pragma unroll
      for (int j = 0; j < NT; ++j)
        acc[j] = __builtin_amdgcn_wmma_f32_16x16x32_bf16(
            false, av, false, cat16(b1lo[j], b1hi[j]), (short)0, acc[j], false, false);
    }
  }
  // epilogue: chunks K-64 (in buf0) and K-32
  load_afrag(arow, K - 32, half, a1lo, a1hi);
#pragma unroll
  for (int j = 0; j < NT; ++j) load_bfrag(brow[j], K - 32, half, b1lo[j], b1hi[j]);
  {
    v16bf av = cat16(a0lo, a0hi);
#pragma unroll
    for (int j = 0; j < NT; ++j)
      acc[j] = __builtin_amdgcn_wmma_f32_16x16x32_bf16(
          false, av, false, cat16(b0lo[j], b0hi[j]), (short)0, acc[j], false, false);
  }
  {
    v16bf av = cat16(a1lo, a1hi);
#pragma unroll
    for (int j = 0; j < NT; ++j)
      acc[j] = __builtin_amdgcn_wmma_f32_16x16x32_bf16(
          false, av, false, cat16(b1lo[j], b1hi[j]), (short)0, acc[j], false, false);
  }

  // D layout: VGPR r, lane -> C[mTile*16 + r + 8*half][ncol*16 + l15]
#pragma unroll
  for (int j = 0; j < NT; ++j) {
#pragma unroll
    for (int r = 0; r < 8; ++r) {
      int mr = mTile * 16 + r + 8 * half;
      if (mr < M)
        C[(size_t)mr * Npad + (ng * NT + j) * 16 + l15] = acc[j][r];
    }
  }
}

// ---------------- attention coefficient kernels ----------------

// one block (HC=256 threads) per node; each 32-lane wave = one head
__global__ void k_alpha1(const float* __restrict__ t1,
                         const float* __restrict__ a_src, const float* __restrict__ a_dst,
                         float* __restrict__ as, float* __restrict__ ad, int H) {
  int n = blockIdx.x;
  int tid = threadIdx.x;
  float v  = t1[(size_t)n * blockDim.x + tid];
  float ps = v * a_src[tid];
  float pd = v * a_dst[tid];
#pragma unroll
  for (int off = 16; off > 0; off >>= 1) {
    ps += __shfl_xor(ps, off, 32);
    pd += __shfl_xor(pd, off, 32);
  }
  if ((tid & 31) == 0) {
    int h = tid >> 5;
    as[(size_t)n * H + h] = ps;
    ad[(size_t)n * H + h] = pd;
  }
}

__global__ void k_alpha2(const float* __restrict__ t2,
                         const float* __restrict__ a_s, const float* __restrict__ a_d,
                         float* __restrict__ as2, float* __restrict__ ad2,
                         int N, int NC, int NCP) {
  int n = blockIdx.x * blockDim.x + threadIdx.x;
  if (n >= N) return;
  float ps = 0.f, pd = 0.f;
  for (int c = 0; c < NC; ++c) {
    float v = t2[(size_t)n * NCP + c];
    ps += v * a_s[c];
    pd += v * a_d[c];
  }
  as2[n] = ps; ad2[n] = pd;
}

// ---------------- layer-1 edge passes (H heads, H a power of two) ----------------

template <int H>
__global__ void k_edge_max1(const int* __restrict__ ei, int E, int Etot,
                            const float* __restrict__ as, const float* __restrict__ ad,
                            float* __restrict__ m) {
  int i = blockIdx.x * blockDim.x + threadIdx.x;
  if (i >= Etot * H) return;
  int e = i / H, h = i & (H - 1);
  int s, d; edge_ends(ei, E, e, s, d);
  float v = as[(size_t)s * H + h] + ad[(size_t)d * H + h];
  v = v > 0.f ? v : NEG_SLOPE * v;
  atomicMaxF(&m[(size_t)d * H + h], v);
}

template <int H>
__global__ void k_edge_sum1(const int* __restrict__ ei, int E, int Etot,
                            const float* __restrict__ as, const float* __restrict__ ad,
                            const float* __restrict__ m, float* __restrict__ ssum) {
  int i = blockIdx.x * blockDim.x + threadIdx.x;
  if (i >= Etot * H) return;
  int e = i / H, h = i & (H - 1);
  int s, d; edge_ends(ei, E, e, s, d);
  float v = as[(size_t)s * H + h] + ad[(size_t)d * H + h];
  v = v > 0.f ? v : NEG_SLOPE * v;
  atomicAdd(&ssum[(size_t)d * H + h], __expf(v - m[(size_t)d * H + h]));
}

// one block (HC threads) per edge: coalesced 1KB gather + 1KB atomic scatter;
// edge endpoints are block-uniform -> scalar loads
template <int H>
__global__ void k_edge_aggr1(const int* __restrict__ ei, int E, int Etot,
                             const float* __restrict__ t1,
                             const float* __restrict__ as, const float* __restrict__ ad,
                             const float* __restrict__ m, const float* __restrict__ ssum,
                             float* __restrict__ agg) {
  int e = blockIdx.x;
  int tid = threadIdx.x;
  int s, d; edge_ends(ei, E, e, s, d);
  int h = tid >> 5;
  float v = as[(size_t)s * H + h] + ad[(size_t)d * H + h];
  v = v > 0.f ? v : NEG_SLOPE * v;
  float alpha = __expf(v - m[(size_t)d * H + h]) / ssum[(size_t)d * H + h];
  atomicAdd(&agg[(size_t)d * blockDim.x + tid],
            t1[(size_t)s * blockDim.x + tid] * alpha);
}

// ---------------- layer-2 edge passes (H = 1) ----------------

__global__ void k_edge_max2(const int* __restrict__ ei, int E, int Etot,
                            const float* __restrict__ as2, const float* __restrict__ ad2,
                            float* __restrict__ m) {
  int e = blockIdx.x * blockDim.x + threadIdx.x;
  if (e >= Etot) return;
  int s, d; edge_ends(ei, E, e, s, d);
  float v = as2[s] + ad2[d];
  v = v > 0.f ? v : NEG_SLOPE * v;
  atomicMaxF(&m[d], v);
}

__global__ void k_edge_sum2(const int* __restrict__ ei, int E, int Etot,
                            const float* __restrict__ as2, const float* __restrict__ ad2,
                            const float* __restrict__ m, float* __restrict__ ssum) {
  int e = blockIdx.x * blockDim.x + threadIdx.x;
  if (e >= Etot) return;
  int s, d; edge_ends(ei, E, e, s, d);
  float v = as2[s] + ad2[d];
  v = v > 0.f ? v : NEG_SLOPE * v;
  atomicAdd(&ssum[d], __expf(v - m[d]));
}

__global__ void k_edge_aggr2(const int* __restrict__ ei, int E, int Etot,
                             const float* __restrict__ t2,
                             const float* __restrict__ as2, const float* __restrict__ ad2,
                             const float* __restrict__ m, const float* __restrict__ ssum,
                             float* __restrict__ agg, int NC, int NCP) {
  int i = blockIdx.x * blockDim.x + threadIdx.x;
  int e = i >> 4, c = i & 15;
  if (e >= Etot) return;
  int s, d; edge_ends(ei, E, e, s, d);
  float v = as2[s] + ad2[d];
  v = v > 0.f ? v : NEG_SLOPE * v;
  float alpha = __expf(v - m[d]) / ssum[d];
  if (c < NC)
    atomicAdd(&agg[(size_t)d * NC + c], t2[(size_t)s * NCP + c] * alpha);
}

// ---------------- elementwise / pooling ----------------

__global__ void k_elu_cvt(const float* __restrict__ agg, const float* __restrict__ b,
                          bf16* __restrict__ hb, int hcMask, size_t total) {
  size_t i = (size_t)blockIdx.x * blockDim.x + threadIdx.x;
  if (i >= total) return;
  float xv = agg[i] + b[i & hcMask];
  hb[i] = (bf16)(xv > 0.f ? xv : expm1f(xv));
}

__global__ void k_pool(const float* __restrict__ agg2, const float* __restrict__ b2,
                       const int* __restrict__ batch,
                       float* __restrict__ pool, float* __restrict__ cnt,
                       int N, int NC) {
  int i = blockIdx.x * blockDim.x + threadIdx.x;
  int n = i >> 4, c = i & 15;
  if (n >= N) return;
  int g = batch[n];
  if (c < NC) atomicAdd(&pool[(size_t)g * NC + c], agg2[(size_t)n * NC + c] + b2[c]);
  if (c == 0) atomicAdd(&cnt[g], 1.0f);
}

__global__ void k_final(const float* __restrict__ pool, const float* __restrict__ cnt,
                        float* __restrict__ out, int total, int NC) {
  int i = blockIdx.x * blockDim.x + threadIdx.x;
  if (i >= total) return;
  out[i] = pool[i] / fmaxf(cnt[i / NC], 1.0f);
}

// ---------------- launch ----------------

extern "C" void kernel_launch(void* const* d_in, const int* in_sizes, int n_in,
                              void* d_out, int out_size, void* d_ws, size_t ws_size,
                              hipStream_t stream) {
  const float* x     = (const float*)d_in[0];
  const float* W1    = (const float*)d_in[1];
  const float* asr1  = (const float*)d_in[2];
  const float* adt1  = (const float*)d_in[3];
  const float* b1    = (const float*)d_in[4];
  const float* W2    = (const float*)d_in[5];
  const float* asr2  = (const float*)d_in[6];
  const float* adt2  = (const float*)d_in[7];
  const float* b2    = (const float*)d_in[8];
  const int*   ei    = (const int*)d_in[9];
  const int*   batch = (const int*)d_in[10];
  float* out = (float*)d_out;
  (void)n_in; (void)ws_size;

  const int N    = in_sizes[10];          // 50000 nodes
  const int E    = in_sizes[9] / 2;       // 800000 edges
  const int F    = in_sizes[0] / N;       // 256
  const int HC   = in_sizes[4];           // H*C = 256
  const int H    = 8;
  const int NC   = in_sizes[8];           // 10
  const int NCP  = 16;                    // padded N for WMMA
  const int G    = out_size / NC;         // 128 graphs
  const int Etot = E + N;                 // + self loops

  // ---- workspace carve-out ----
  char* wsp = (char*)d_ws;
  auto take = [&](size_t bytes) -> char* {
    char* p = wsp; wsp += (bytes + 255) & ~(size_t)255; return p;
  };
  float* t1   = (float*)take((size_t)N * HC * 4);     // x @ W1
  float* agg1 = (float*)take((size_t)N * HC * 4);     // layer-1 aggregation
  bf16*  xb   = (bf16*) take((size_t)N * F  * 2);     // x bf16, reused as h bf16
  bf16*  w1t  = (bf16*) take((size_t)HC * F * 2);     // W1^T bf16
  bf16*  w2t  = (bf16*) take((size_t)NCP * HC * 2);   // W2^T bf16 (padded)
  float* as1  = (float*)take((size_t)N * H * 4);
  float* ad1  = (float*)take((size_t)N * H * 4);
  float* m1   = (float*)take((size_t)N * H * 4);
  float* s1   = (float*)take((size_t)N * H * 4);
  float* t2   = (float*)take((size_t)N * NCP * 4);    // h @ W2 (padded)
  float* as2  = (float*)take((size_t)N * 4);
  float* ad2  = (float*)take((size_t)N * 4);
  float* m2   = (float*)take((size_t)N * 4);
  float* s2   = (float*)take((size_t)N * 4);
  float* agg2 = (float*)take((size_t)N * NC * 4);
  float* pool = (float*)take((size_t)G * NC * 4);
  float* cnt  = (float*)take((size_t)G * 4);

  const int T = 256;
  auto nb = [](size_t work, int t) { return (int)((work + t - 1) / t); };

  // ---- init accumulators (graph-capture-safe) ----
  hipMemsetAsync(agg1, 0, (size_t)N * HC * 4, stream);
  hipMemsetAsync(s1,   0, (size_t)N * H  * 4, stream);
  hipMemsetAsync(s2,   0, (size_t)N * 4,      stream);
  hipMemsetAsync(agg2, 0, (size_t)N * NC * 4, stream);
  hipMemsetAsync(pool, 0, (size_t)G * NC * 4, stream);
  hipMemsetAsync(cnt,  0, (size_t)G * 4,      stream);
  k_fill<<<nb((size_t)N * H, T), T, 0, stream>>>(m1, -INFINITY, (size_t)N * H);
  k_fill<<<nb((size_t)N, T), T, 0, stream>>>(m2, -INFINITY, (size_t)N);

  // ---- precision conversion ----
  k_cvt_bf16<<<nb((size_t)N * F, T), T, 0, stream>>>(x, xb, (size_t)N * F);
  k_transpose_cvt<<<nb((size_t)HC * F, T), T, 0, stream>>>(W1, w1t, F, HC, HC);
  k_transpose_cvt<<<nb((size_t)NCP * HC, T), T, 0, stream>>>(W2, w2t, HC, NC, NCP);

  // ---- layer 1: GEMM (WMMA bf16), attention coeffs, softmax, aggregate ----
  {
    int mTiles = (N + 15) / 16;                  // 3125
    int waves  = mTiles * (HC / (16 * 4));       // NT=4 -> 12500 waves
    k_gemm_wmma_bf16<4><<<nb((size_t)waves * 32, T), T, 0, stream>>>(
        xb, w1t, t1, N, F, HC, waves);
  }
  k_alpha1<<<N, HC, 0, stream>>>(t1, asr1, adt1, as1, ad1, H);
  k_edge_max1<8><<<nb((size_t)Etot * H, T), T, 0, stream>>>(ei, E, Etot, as1, ad1, m1);
  k_edge_sum1<8><<<nb((size_t)Etot * H, T), T, 0, stream>>>(ei, E, Etot, as1, ad1, m1, s1);
  k_edge_aggr1<8><<<Etot, HC, 0, stream>>>(ei, E, Etot, t1, as1, ad1, m1, s1, agg1);

  // ---- ELU(+b1) and bf16 conversion for layer 2 (reuse xb) ----
  k_elu_cvt<<<nb((size_t)N * HC, T), T, 0, stream>>>(agg1, b1, xb, HC - 1, (size_t)N * HC);

  // ---- layer 2: GEMM (WMMA bf16, Npad=16), attention, softmax, aggregate ----
  {
    int mTiles = (N + 15) / 16;
    int waves  = mTiles;                         // NT=1, one 16-wide N tile
    k_gemm_wmma_bf16<1><<<nb((size_t)waves * 32, T), T, 0, stream>>>(
        xb, w2t, t2, N, HC, NCP, waves);
  }
  k_alpha2<<<nb((size_t)N, T), T, 0, stream>>>(t2, asr2, adt2, as2, ad2, N, NC, NCP);
  k_edge_max2<<<nb((size_t)Etot, T), T, 0, stream>>>(ei, E, Etot, as2, ad2, m2);
  k_edge_sum2<<<nb((size_t)Etot, T), T, 0, stream>>>(ei, E, Etot, as2, ad2, m2, s2);
  k_edge_aggr2<<<nb((size_t)Etot * 16, T), T, 0, stream>>>(
      ei, E, Etot, t2, as2, ad2, m2, s2, agg2, NC, NCP);

  // ---- global mean pool (+b2) ----
  k_pool<<<nb((size_t)N * 16, T), T, 0, stream>>>(agg2, b2, batch, pool, cnt, N, NC);
  k_final<<<nb((size_t)G * NC, T), T, 0, stream>>>(pool, cnt, out, G * NC, NC);
}
